// EEGConvNetMiniLSTM_7112465842811
// MI455X (gfx1250) — compile-verified, matrix operand order checked by reading, not run on hardware
//
#include <hip/hip_runtime.h>
#include <stddef.h>

// ---------------------------------------------------------------------------
// MI455X (gfx1250) implementation of EEGConvNetMiniLSTM.
//
//   prep:   combined LSTM biases; ALL GEMM operands packed to bf16 once
//   gemm:   bf16 WMMA GEMM (v_wmma_f32_16x16x32_bf16), 32x64 tile / wave,
//           fragments loaded directly as bf16 via b128 loads; all four
//           B-fragments preloaded per k-step so waits are staged and loads
//           overlap WMMA execution
//   lstm:   16 persistent workgroups, register-resident W_hh slices,
//           double-buffered global h exchange + software global barrier
//   gcn:    WMMA projection + f32 atomic edge scatter + column batchnorm
//   head:   pooling atomics + per-graph tiny MLP
// ---------------------------------------------------------------------------

typedef __attribute__((ext_vector_type(16))) __bf16    v16bf;
typedef __attribute__((ext_vector_type(8)))  float     v8f;
typedef __attribute__((ext_vector_type(4)))  unsigned  u32x4;

union ABFrag { v16bf v; u32x4 q[2]; };

__device__ __forceinline__ unsigned short f2bf(float x) {
  unsigned u = __float_as_uint(x);
  u += 0x7FFFu + ((u >> 16) & 1u);          // round-to-nearest-even
  return (unsigned short)(u >> 16);
}
__device__ __forceinline__ float bflo(unsigned u) { return __uint_as_float(u << 16); }
__device__ __forceinline__ float bfhi(unsigned u) { return __uint_as_float(u & 0xFFFF0000u); }
__device__ __forceinline__ float lreluf(float x)  { return x > 0.f ? x : 0.01f * x; }
__device__ __forceinline__ float sigmf(float x)   { return 1.f / (1.f + __expf(-x)); }

// ---------------------------------------------------------------------------
// Utility kernels (workspace init, deterministic across graph replays)
// ---------------------------------------------------------------------------
__global__ void zero_f32(float* p, int n) {
  int i = blockIdx.x * blockDim.x + threadIdx.x;
  if (i < n) p[i] = 0.f;
}
__global__ void zero_u32(unsigned* p, int n) {
  int i = blockIdx.x * blockDim.x + threadIdx.x;
  if (i < n) p[i] = 0u;
}
__global__ void vec_add(const float* __restrict__ a, const float* __restrict__ b,
                        float* __restrict__ d, int n) {
  int i = blockIdx.x * blockDim.x + threadIdx.x;
  if (i < n) d[i] = a[i] + b[i];
}
// pack pairs of f32 into one dword of 2x bf16 (lo = even index, hi = odd)
__global__ void pack_bf16(const float* __restrict__ srcp, unsigned* __restrict__ dstp,
                          int npairs) {
  int i = blockIdx.x * blockDim.x + threadIdx.x;
  if (i < npairs)
    dstp[i] = (unsigned)f2bf(srcp[2 * i]) | ((unsigned)f2bf(srcp[2 * i + 1]) << 16);
}

// ---------------------------------------------------------------------------
// bf16 WMMA GEMM:  C[M,N] = act( A[M,K] @ B[N,K]^T + bias )
// A and B are pre-packed bf16 (2 elems / dword), row-major, K contiguous.
// Per the CDNA5 ISA 7.12.2 16-bit striping, each lane's fragment is two runs
// of 8 contiguous bf16 (k = klo+0..7 and k = 16+klo+0..7, klo = 0|8), so a
// fragment is exactly two aligned 128-bit loads -- zero conversion VALU.
// Wave tile = 32(M) x 64(N): 2 A-frags x 4 B-frags -> 8 WMMAs per k-step,
// with each B-frag reused across both M-tiles. Block = 8 waves = 64M x 256N.
// Fast path: all fragments preloaded -> staged s_wait_loadcnt, loads overlap
// WMMA execution within the wave.
// ---------------------------------------------------------------------------
__global__ __launch_bounds__(256)
void gemm_bf16_wmma(const unsigned* __restrict__ Abf, const unsigned* __restrict__ Bbf,
                    const float* __restrict__ bias, float* __restrict__ C,
                    int M, int N, int K, int act)
{
  const int lane    = threadIdx.x & 31;
  const int w       = threadIdx.x >> 5;
  const int m0      = blockIdx.y * 64 + (w & 1) * 32;
  const int n0base  = blockIdx.x * 256 + (w >> 1) * 64;
  if (m0 >= M || n0base >= N) return;

  const int l15     = lane & 15;
  const int kloD    = (lane < 16) ? 0 : 4;   // dword offset of lane's K-run
  const int halfsel = (lane < 16) ? 0 : 8;   // C/D: lanes 16-31 hold M+8
  const int Kd      = K >> 1;                // dwords per row

  v8f zero8 = {0.f, 0.f, 0.f, 0.f, 0.f, 0.f, 0.f, 0.f};
  v8f acc[2][4] = {{zero8, zero8, zero8, zero8}, {zero8, zero8, zero8, zero8}};

  const unsigned* Arow0 = Abf + (size_t)(m0 + l15) * Kd + kloD;
  const unsigned* Arow1 = Arow0 + (size_t)16 * Kd;
  const unsigned* Brow[4];
#pragma unroll
  for (int i = 0; i < 4; ++i)
    Brow[i] = Bbf + (size_t)(n0base + 16 * i + l15) * Kd + kloD;

  if (n0base + 64 <= N) {
    // ---- fast path: all four 16-wide N tiles valid, no guards ------------
    for (int kd = 0; kd < Kd; kd += 16) {    // 16 dwords = 32 K per step
      ABFrag a0, a1, b0, b1, b2, b3;
      a0.q[0] = *(const u32x4*)(Arow0 + kd);
      a0.q[1] = *(const u32x4*)(Arow0 + kd + 8);
      a1.q[0] = *(const u32x4*)(Arow1 + kd);
      a1.q[1] = *(const u32x4*)(Arow1 + kd + 8);
      b0.q[0] = *(const u32x4*)(Brow[0] + kd);
      b0.q[1] = *(const u32x4*)(Brow[0] + kd + 8);
      b1.q[0] = *(const u32x4*)(Brow[1] + kd);
      b1.q[1] = *(const u32x4*)(Brow[1] + kd + 8);
      b2.q[0] = *(const u32x4*)(Brow[2] + kd);
      b2.q[1] = *(const u32x4*)(Brow[2] + kd + 8);
      b3.q[0] = *(const u32x4*)(Brow[3] + kd);
      b3.q[1] = *(const u32x4*)(Brow[3] + kd + 8);
      if (kd + 16 < Kd) {
        __builtin_prefetch(Arow0 + kd + 16, 0, 3);
        __builtin_prefetch(Arow1 + kd + 16, 0, 3);
      }
      acc[0][0] = __builtin_amdgcn_wmma_f32_16x16x32_bf16(
          false, a0.v, false, b0.v, (short)0, acc[0][0], false, false);
      acc[1][0] = __builtin_amdgcn_wmma_f32_16x16x32_bf16(
          false, a1.v, false, b0.v, (short)0, acc[1][0], false, false);
      acc[0][1] = __builtin_amdgcn_wmma_f32_16x16x32_bf16(
          false, a0.v, false, b1.v, (short)0, acc[0][1], false, false);
      acc[1][1] = __builtin_amdgcn_wmma_f32_16x16x32_bf16(
          false, a1.v, false, b1.v, (short)0, acc[1][1], false, false);
      acc[0][2] = __builtin_amdgcn_wmma_f32_16x16x32_bf16(
          false, a0.v, false, b2.v, (short)0, acc[0][2], false, false);
      acc[1][2] = __builtin_amdgcn_wmma_f32_16x16x32_bf16(
          false, a1.v, false, b2.v, (short)0, acc[1][2], false, false);
      acc[0][3] = __builtin_amdgcn_wmma_f32_16x16x32_bf16(
          false, a0.v, false, b3.v, (short)0, acc[0][3], false, false);
      acc[1][3] = __builtin_amdgcn_wmma_f32_16x16x32_bf16(
          false, a1.v, false, b3.v, (short)0, acc[1][3], false, false);
    }
  } else {
    // ---- guarded path: N tile < 64 wide (the N=32 projection) ------------------
    for (int kd = 0; kd < Kd; kd += 16) {
      ABFrag a0, a1;
      a0.q[0] = *(const u32x4*)(Arow0 + kd);
      a0.q[1] = *(const u32x4*)(Arow0 + kd + 8);
      a1.q[0] = *(const u32x4*)(Arow1 + kd);
      a1.q[1] = *(const u32x4*)(Arow1 + kd + 8);
#pragma unroll
      for (int i = 0; i < 4; ++i) {
        if (n0base + 16 * i < N) {
          ABFrag b;
          b.q[0] = *(const u32x4*)(Brow[i] + kd);
          b.q[1] = *(const u32x4*)(Brow[i] + kd + 8);
          acc[0][i] = __builtin_amdgcn_wmma_f32_16x16x32_bf16(
              false, a0.v, false, b.v, (short)0, acc[0][i], false, false);
          acc[1][i] = __builtin_amdgcn_wmma_f32_16x16x32_bf16(
              false, a1.v, false, b.v, (short)0, acc[1][i], false, false);
        }
      }
    }
  }

#pragma unroll
  for (int mi = 0; mi < 2; ++mi) {
#pragma unroll
    for (int i = 0; i < 4; ++i) {
      int n0 = n0base + 16 * i;
      if (n0 < N) {
        int ncol = n0 + l15;
        float bv = bias ? bias[ncol] : 0.f;
#pragma unroll
        for (int e = 0; e < 8; ++e) {
          int row = m0 + 16 * mi + halfsel + e;
          float v = acc[mi][i][e] + bv;
          if (act) v = lreluf(v);
          C[(size_t)row * N + ncol] = v;
        }
      }
    }
  }
}

// ---------------------------------------------------------------------------
// Persistent LSTM recurrence. 16 blocks; block b owns hidden slice
// [b*HS, b*HS+HS). Its 4*HS gate rows of W_hh live bf16-packed in registers
// for the whole 4096-step loop. h is exchanged through a double-buffered
// global array; steps are separated by a software global barrier.
// ---------------------------------------------------------------------------
template <int H, int HS, int TPR, int KC>
__global__ __launch_bounds__(4 * HS * TPR)
void lstm_rec(const float* __restrict__ gx,       // [T, 4H] input projection
              const unsigned* __restrict__ whh,   // [4H, H/2] bf16-pairs
              float* __restrict__ hglob,          // [2, H] exchange buffers
              float* __restrict__ out,            // [T, H] = lrelu(h_t)
              unsigned* __restrict__ barcnt, int T, int nblocks)
{
  constexpr int R   = 4 * HS;
  constexpr int TPB = R * TPR;
  static_assert(TPR * KC == H, "bad split");

  const int tid   = threadIdx.x;
  const int b     = blockIdx.x;
  const int r     = tid / TPR;        // local gate row 0..R-1
  const int kseg  = tid % TPR;
  const int q     = r / HS;           // gate 0..3 (i,f,g,o)
  const int u     = r % HS;           // hidden unit within slice
  const int grow  = q * H + b * HS + u;
  const int kbase = kseg * KC;

  unsigned wreg[KC / 2];
  {
    const unsigned* wp = whh + (size_t)grow * (H / 2) + (kbase >> 1);
#pragma unroll
    for (int j = 0; j < KC / 2; ++j) wreg[j] = wp[j];
  }

  __shared__ float h_sh[H];
  __shared__ float part[TPB];
  __shared__ float gates[R];
  __shared__ float c_sh[HS];
  if (tid < HS) c_sh[tid] = 0.f;

  for (int t = 0; t < T; ++t) {
    const int rbuf = t & 1;
    const int wbuf = rbuf ^ 1;

    for (int i = tid; i < H; i += TPB)
      h_sh[i] = __hip_atomic_load(&hglob[rbuf * H + i], __ATOMIC_RELAXED,
                                  __HIP_MEMORY_SCOPE_AGENT);
    __syncthreads();

    float s = 0.f;
#pragma unroll
    for (int j = 0; j < KC / 2; ++j) {
      unsigned wv = wreg[j];
      s += bflo(wv) * h_sh[kbase + 2 * j] + bfhi(wv) * h_sh[kbase + 2 * j + 1];
    }
    part[tid] = s;
    __syncthreads();

    if (tid < R) {
      float g = gx[(size_t)t * (4 * H) + (tid / HS) * H + b * HS + (tid % HS)];
#pragma unroll
      for (int j = 0; j < TPR; ++j) g += part[tid * TPR + j];
      gates[tid] = g;
    }
    __syncthreads();

    if (tid < HS) {
      float iv = sigmf(gates[tid]);
      float fv = sigmf(gates[HS + tid]);
      float gv = tanhf(gates[2 * HS + tid]);
      float ov = sigmf(gates[3 * HS + tid]);
      float c  = fv * c_sh[tid] + iv * gv;
      float h  = ov * tanhf(c);
      c_sh[tid] = c;
      int hg = b * HS + tid;
      __hip_atomic_store(&hglob[wbuf * H + hg], h, __ATOMIC_RELAXED,
                         __HIP_MEMORY_SCOPE_AGENT);
      out[(size_t)t * H + hg] = lreluf(h);   // activation only on the emitted seq
    }
    __threadfence();
    __syncthreads();
    if (tid == 0) {                          // global barrier across 16 blocks
      __hip_atomic_fetch_add(barcnt, 1u, __ATOMIC_RELEASE, __HIP_MEMORY_SCOPE_AGENT);
      const unsigned target = (unsigned)nblocks * (unsigned)(t + 1);
      while (__hip_atomic_load(barcnt, __ATOMIC_ACQUIRE, __HIP_MEMORY_SCOPE_AGENT) < target)
        __builtin_amdgcn_s_sleep(1);
    }
    __syncthreads();
  }
}

// ---------------------------------------------------------------------------
// GCN edge scatter:  agg[dst] += ew * xw[src]   (one thread per edge*channel)
// ---------------------------------------------------------------------------
__global__ void gcn_scatter(const float* __restrict__ xw, const int* __restrict__ srcI,
                            const int* __restrict__ dstI, const float* __restrict__ ew,
                            float* __restrict__ agg, int E, int C)
{
  int idx = blockIdx.x * blockDim.x + threadIdx.x;
  if (idx >= E * C) return;
  int e = idx / C, c = idx - e * C;
  atomicAdd(&agg[(size_t)dstI[e] * C + c], ew[e] * xw[(size_t)srcI[e] * C + c]);
}

// ---------------------------------------------------------------------------
// Training-mode BatchNorm1d over node dim + gcn bias + leaky relu.
// One block per channel.
// ---------------------------------------------------------------------------
__global__ __launch_bounds__(256)
void bn_lrelu(const float* __restrict__ x, const float* __restrict__ convb,
              const float* __restrict__ gamma, const float* __restrict__ beta,
              float* __restrict__ y, int Nrows, int C)
{
  int c = blockIdx.x;
  __shared__ float s1[256], s2[256];
  float a = 0.f, b2 = 0.f;
  float cb = convb[c];
  for (int r = threadIdx.x; r < Nrows; r += 256) {
    float v = x[(size_t)r * C + c] + cb;
    a += v; b2 += v * v;
  }
  s1[threadIdx.x] = a; s2[threadIdx.x] = b2;
  __syncthreads();
  for (int o = 128; o > 0; o >>= 1) {
    if (threadIdx.x < o) {
      s1[threadIdx.x] += s1[threadIdx.x + o];
      s2[threadIdx.x] += s2[threadIdx.x + o];
    }
    __syncthreads();
  }
  float m   = s1[0] / (float)Nrows;
  float var = s2[0] / (float)Nrows - m * m;       // biased variance
  float sc  = gamma[c] * rsqrtf(var + 1e-5f);
  float sh  = beta[c];
  for (int r = threadIdx.x; r < Nrows; r += 256) {
    float v = (x[(size_t)r * C + c] + cb - m) * sc + sh;
    y[(size_t)r * C + c] = lreluf(v);
  }
}

// global_add_pool
__global__ void pool_add(const float* __restrict__ x, const int* __restrict__ batch,
                         float* __restrict__ pooled, int Nn, int C)
{
  int idx = blockIdx.x * blockDim.x + threadIdx.x;
  if (idx >= Nn * C) return;
  int n = idx / C, c = idx - n * C;
  atomicAdd(&pooled[(size_t)batch[n] * C + c], x[idx]);
}

// tiny per-graph MLP head: 32 -> 16 -> 8 -> 2, lrelu each
__global__ void head_mlp(const float* __restrict__ pooled,
                         const float* __restrict__ w2, const float* __restrict__ b2v,
                         const float* __restrict__ w3, const float* __restrict__ b3v,
                         const float* __restrict__ w4, const float* __restrict__ b4v,
                         float* __restrict__ outp, int G)
{
  int g = blockIdx.x * blockDim.x + threadIdx.x;
  if (g >= G) return;
  float in[32];
#pragma unroll
  for (int i = 0; i < 32; ++i) in[i] = pooled[g * 32 + i];
  float h1[16];
#pragma unroll
  for (int o = 0; o < 16; ++o) {
    float s = b2v[o];
#pragma unroll
    for (int i = 0; i < 32; ++i) s += w2[o * 32 + i] * in[i];
    h1[o] = lreluf(s);
  }
  float h2[8];
#pragma unroll
  for (int o = 0; o < 8; ++o) {
    float s = b3v[o];
#pragma unroll
    for (int i = 0; i < 16; ++i) s += w3[o * 16 + i] * h1[i];
    h2[o] = lreluf(s);
  }
#pragma unroll
  for (int o = 0; o < 2; ++o) {
    float s = b4v[o];
#pragma unroll
    for (int i = 0; i < 8; ++i) s += w4[o * 8 + i] * h2[i];
    outp[g * 2 + o] = lreluf(s);
  }
}

// ---------------------------------------------------------------------------
// Host orchestration
// ---------------------------------------------------------------------------
extern "C" void kernel_launch(void* const* d_in, const int* in_sizes, int n_in,
                              void* d_out, int out_size, void* d_ws, size_t ws_size,
                              hipStream_t stream)
{
  (void)in_sizes; (void)n_in; (void)out_size; (void)ws_size;

  const float* x      = (const float*)d_in[0];
  const int*   ei     = (const int*)d_in[1];
  const float* ew     = (const float*)d_in[2];
  const int*   batch  = (const int*)d_in[3];
  const float* W_ih1  = (const float*)d_in[4];
  const float* W_hh1  = (const float*)d_in[5];
  const float* b_ih1  = (const float*)d_in[6];
  const float* b_hh1  = (const float*)d_in[7];
  const float* W_ih2  = (const float*)d_in[8];
  const float* W_hh2  = (const float*)d_in[9];
  const float* b_ih2  = (const float*)d_in[10];
  const float* b_hh2  = (const float*)d_in[11];
  const float* fc1_w  = (const float*)d_in[12];
  const float* fc1_b  = (const float*)d_in[13];
  const float* gcn1_w = (const float*)d_in[14];
  const float* gcn1_b = (const float*)d_in[15];
  const float* bn1_g  = (const float*)d_in[16];
  const float* bn1_b  = (const float*)d_in[17];
  const float* gcn2_w = (const float*)d_in[18];
  const float* gcn2_b = (const float*)d_in[19];
  const float* bn2_g  = (const float*)d_in[20];
  const float* bn2_b  = (const float*)d_in[21];
  const float* fc2_w  = (const float*)d_in[22];
  const float* fc2_b  = (const float*)d_in[23];
  const float* fc3_w  = (const float*)d_in[24];
  const float* fc3_b  = (const float*)d_in[25];
  const float* fc4_w  = (const float*)d_in[26];
  const float* fc4_b  = (const float*)d_in[27];

  const int N = 4096, E = 32768, G = 512;
  const int* srcI = ei;
  const int* dstI = ei + E;

  // workspace carve-up (64-float = 256B aligned slots)
  float* wsf = (float*)d_ws;
  size_t off = 0;
  auto alloc = [&](size_t nfl) { size_t o = off; off += (nfl + 63) & ~(size_t)63; return o; };
  // f32 activations
  float*    gx1    = wsf + alloc((size_t)N * 2048);
  float*    out1   = wsf + alloc((size_t)N * 512);
  float*    gx2    = wsf + alloc((size_t)N * 1024);
  float*    out2   = wsf + alloc((size_t)N * 256);
  float*    fc1o   = wsf + alloc((size_t)N * 128);
  float*    xw1    = wsf + alloc((size_t)N * 64);
  float*    agg1   = wsf + alloc((size_t)N * 64);
  float*    bn1o   = wsf + alloc((size_t)N * 64);
  float*    xw2    = wsf + alloc((size_t)N * 32);
  float*    agg2   = wsf + alloc((size_t)N * 32);
  float*    bn2o   = wsf + alloc((size_t)N * 32);
  float*    pooled = wsf + alloc((size_t)G * 32);
  float*    b1c    = wsf + alloc(2048);
  float*    b2c    = wsf + alloc(1024);
  // bf16-packed GEMM operands (elems/2 dwords each)
  unsigned* xbf    = (unsigned*)(wsf + alloc((size_t)N * 1280 / 2));
  unsigned* wih1bf = (unsigned*)(wsf + alloc((size_t)2048 * 1280 / 2));
  unsigned* wih2bf = (unsigned*)(wsf + alloc((size_t)1024 * 512 / 2));
  unsigned* fc1wbf = (unsigned*)(wsf + alloc((size_t)128 * 256 / 2));
  unsigned* g1wbf  = (unsigned*)(wsf + alloc((size_t)64 * 128 / 2));
  unsigned* g2wbf  = (unsigned*)(wsf + alloc((size_t)32 * 64 / 2));
  unsigned* out1bf = (unsigned*)(wsf + alloc((size_t)N * 512 / 2));
  unsigned* out2bf = (unsigned*)(wsf + alloc((size_t)N * 256 / 2));
  unsigned* fc1obf = (unsigned*)(wsf + alloc((size_t)N * 128 / 2));
  unsigned* bn1obf = (unsigned*)(wsf + alloc((size_t)N * 64 / 2));
  // LSTM recurrent weights (bf16) + exchange + barrier
  unsigned* whh1p  = (unsigned*)(wsf + alloc(2048 * 512 / 2));
  unsigned* whh2p  = (unsigned*)(wsf + alloc(1024 * 256 / 2));
  float*    hg1    = wsf + alloc(2 * 512);
  float*    hg2    = wsf + alloc(2 * 256);
  unsigned* barcnt = (unsigned*)(wsf + alloc(64));

  auto PAIRS = [](size_t nelem) { return (int)(nelem / 2); };
  auto GRID1 = [](int n) { return (n + 255) / 256; };

  // ---- prep --------------------------------------------------------------
  vec_add<<<8, 256, 0, stream>>>(b_ih1, b_hh1, b1c, 2048);
  vec_add<<<4, 256, 0, stream>>>(b_ih2, b_hh2, b2c, 1024);
  pack_bf16<<<GRID1(PAIRS((size_t)N * 1280)), 256, 0, stream>>>(x, xbf, PAIRS((size_t)N * 1280));
  pack_bf16<<<GRID1(PAIRS((size_t)2048 * 1280)), 256, 0, stream>>>(W_ih1, wih1bf, PAIRS((size_t)2048 * 1280));
  pack_bf16<<<GRID1(PAIRS((size_t)1024 * 512)), 256, 0, stream>>>(W_ih2, wih2bf, PAIRS((size_t)1024 * 512));
  pack_bf16<<<GRID1(PAIRS((size_t)128 * 256)), 256, 0, stream>>>(fc1_w, fc1wbf, PAIRS((size_t)128 * 256));
  pack_bf16<<<GRID1(PAIRS((size_t)64 * 128)), 256, 0, stream>>>(gcn1_w, g1wbf, PAIRS((size_t)64 * 128));
  pack_bf16<<<GRID1(PAIRS((size_t)32 * 64)), 256, 0, stream>>>(gcn2_w, g2wbf, PAIRS((size_t)32 * 64));
  pack_bf16<<<GRID1(PAIRS((size_t)2048 * 512)), 256, 0, stream>>>(W_hh1, whh1p, PAIRS((size_t)2048 * 512));
  pack_bf16<<<GRID1(PAIRS((size_t)1024 * 256)), 256, 0, stream>>>(W_hh2, whh2p, PAIRS((size_t)1024 * 256));
  zero_f32<<<4, 256, 0, stream>>>(hg1, 2 * 512);
  zero_f32<<<2, 256, 0, stream>>>(hg2, 2 * 256);
  zero_f32<<<GRID1(N * 64), 256, 0, stream>>>(agg1, N * 64);
  zero_f32<<<GRID1(N * 32), 256, 0, stream>>>(agg2, N * 32);
  zero_f32<<<GRID1(G * 32), 256, 0, stream>>>(pooled, G * 32);
  zero_u32<<<1, 64, 0, stream>>>(barcnt, 64);

  // ---- LSTM 1: gx1 = x @ W_ih1.T + (b_ih1+b_hh1);  recurrence ------------
  gemm_bf16_wmma<<<dim3(8, 64), 256, 0, stream>>>(xbf, wih1bf, b1c, gx1,
                                                  N, 2048, 1280, 0);
  lstm_rec<512, 32, 4, 128><<<16, 512, 0, stream>>>(gx1, whh1p, hg1, out1,
                                                    barcnt, N, 16);
  pack_bf16<<<GRID1(PAIRS((size_t)N * 512)), 256, 0, stream>>>(out1, out1bf, PAIRS((size_t)N * 512));

  // ---- LSTM 2 ------------------------------------------------------------
  gemm_bf16_wmma<<<dim3(4, 64), 256, 0, stream>>>(out1bf, wih2bf, b2c, gx2,
                                                  N, 1024, 512, 0);
  lstm_rec<256, 16, 8, 32><<<16, 512, 0, stream>>>(gx2, whh2p, hg2, out2,
                                                   barcnt + 32, N, 16);
  pack_bf16<<<GRID1(PAIRS((size_t)N * 256)), 256, 0, stream>>>(out2, out2bf, PAIRS((size_t)N * 256));

  // ---- fc1 + lrelu -------------------------------------------------------
  gemm_bf16_wmma<<<dim3(1, 64), 256, 0, stream>>>(out2bf, fc1wbf, fc1_b, fc1o,
                                                  N, 128, 256, 1);
  pack_bf16<<<GRID1(PAIRS((size_t)N * 128)), 256, 0, stream>>>(fc1o, fc1obf, PAIRS((size_t)N * 128));

  // ---- GCN layer 1 -------------------------------------------------------
  gemm_bf16_wmma<<<dim3(1, 64), 256, 0, stream>>>(fc1obf, g1wbf, nullptr, xw1,
                                                  N, 64, 128, 0);
  gcn_scatter<<<GRID1(E * 64), 256, 0, stream>>>(xw1, srcI, dstI, ew, agg1, E, 64);
  bn_lrelu<<<64, 256, 0, stream>>>(agg1, gcn1_b, bn1_g, bn1_b, bn1o, N, 64);
  pack_bf16<<<GRID1(PAIRS((size_t)N * 64)), 256, 0, stream>>>(bn1o, bn1obf, PAIRS((size_t)N * 64));

  // ---- GCN layer 2 -------------------------------------------------------
  gemm_bf16_wmma<<<dim3(1, 64), 256, 0, stream>>>(bn1obf, g2wbf, nullptr, xw2,
                                                  N, 32, 64, 0);
  gcn_scatter<<<GRID1(E * 32), 256, 0, stream>>>(xw2, srcI, dstI, ew, agg2, E, 32);
  bn_lrelu<<<32, 256, 0, stream>>>(agg2, gcn2_b, bn2_g, bn2_b, bn2o, N, 32);

  // ---- pool + head -------------------------------------------------------
  pool_add<<<GRID1(N * 32), 256, 0, stream>>>(bn2o, batch, pooled, N, 32);
  head_mlp<<<2, 256, 0, stream>>>(pooled, fc2_w, fc2_b, fc3_w, fc3_b,
                                  fc4_w, fc4_b, (float*)d_out, G);
}
